// tinyBlock_17566416240908
// MI455X (gfx1250) — matrix operand order repaired and checked by reading, not compiled, not run on hardware
//
#include <hip/hip_runtime.h>
#include <hip/hip_bf16.h>
#include <math.h>

// ---------------------------------------------------------------------------
// Types for CDNA5 WMMA (wave32): v_wmma_f32_16x16x32_bf16
// ---------------------------------------------------------------------------
typedef float  v8f   __attribute__((ext_vector_type(8)));
typedef int    v4i   __attribute__((ext_vector_type(4)));
typedef __bf16 v8bf  __attribute__((ext_vector_type(8)));
typedef __bf16 v16bf __attribute__((ext_vector_type(16)));

union BF16Frag { v16bf f; v8bf h[2]; };

#define AS1 __attribute__((address_space(1)))
#define AS3 __attribute__((address_space(3)))

// ---------------------------------------------------------------------------
// CDNA5 async global->LDS copy (ASYNCcnt-tracked), 16B per lane per issue.
// Builtin signature (probed): (v4i AS1*, v4i AS3*, imm offset, imm cpol)
// ---------------------------------------------------------------------------
__device__ __forceinline__ void async_copy_b128(void* lds_dst, const void* gsrc)
{
#if __has_builtin(__builtin_amdgcn_global_load_async_to_lds_b128)
    __builtin_amdgcn_global_load_async_to_lds_b128(
        (AS1 v4i*)gsrc, (AS3 v4i*)lds_dst, 0, 0);
#else
    unsigned lo = (unsigned)(size_t)(AS3 void*)lds_dst;       // 32-bit LDS addr
    unsigned long long ga = (unsigned long long)(size_t)gsrc; // 64-bit global
    asm volatile("global_load_async_to_lds_b128 %0, %1, off"
                 :: "v"(lo), "v"(ga) : "memory");
#endif
}

template <int N>
__device__ __forceinline__ void wait_asynccnt()
{
#if __has_builtin(__builtin_amdgcn_s_wait_asynccnt)
    __builtin_amdgcn_s_wait_asynccnt(N);
#else
    asm volatile("s_wait_asynccnt %0" :: "n"(N) : "memory");
#endif
}

// ---------------------------------------------------------------------------
// Generic tiled GEMM:  C[M][N] = epilogue( A[M][K](bf16) * Bt[N][K](bf16) )
// BM=256, BN=128, BK=32; 256 threads = 8 waves; each wave -> 64(M) x 64(N)
// (4x4 WMMA tiles, 16 WMMAs per K-step, 1.0 LDS b128 loads per WMMA).
// Double-buffered LDS, async global->LDS copies pipelined against WMMA.
// A row stride = K, Bt row stride = K, C/res row stride = N.
// M % 256 == 0, N % 128 == 0, K % 32 == 0 (guaranteed by caller).
// ---------------------------------------------------------------------------
template <bool HAS_BIAS, bool HAS_RES, bool GELU_ACT, bool OUT_BF16>
__global__ __launch_bounds__(256) void k_gemm_bf16(
    const __bf16* __restrict__ A, const __bf16* __restrict__ Bt,
    const float* __restrict__ bias, const float* __restrict__ res,
    float* __restrict__ outF, __bf16* __restrict__ outB,
    int M, int N, int K)
{
    __shared__ __bf16 As[2][256][40];  // 32 K-elems + pad (80B rows, 16B aligned)
    __shared__ __bf16 Bs[2][128][40];

    const int tid  = threadIdx.x;
    const int wave = tid >> 5;
    const int lane = tid & 31;
    const int l15  = lane & 15;
    const int hi   = lane >> 4;          // 0: lanes 0-15, 1: lanes 16-31
    const int wm   = wave & 3;           // 0..3  -> 64-row slabs
    const int wn   = wave >> 2;          // 0..1  -> 64-col slabs
    const int m0   = blockIdx.y * 256;
    const int n0   = blockIdx.x * 128;

    // staging: thread covers A rows rS+{0,64,128,192}, B rows rS+{0,64},
    // all at a fixed 8-elem (16B) K chunk.
    const int rS = tid >> 2;             // 0..63
    const int kc = (tid & 3) * 8;        // 0,8,16,24
    const __bf16* gA0 = A  + (size_t)(m0 + rS) * K + kc;
    const __bf16* gA1 = gA0 + (size_t)64  * K;
    const __bf16* gA2 = gA0 + (size_t)128 * K;
    const __bf16* gA3 = gA0 + (size_t)192 * K;
    const __bf16* gB0 = Bt + (size_t)(n0 + rS) * K + kc;
    const __bf16* gB1 = gB0 + (size_t)64 * K;

    auto issue_tile = [&](int k0, int bi) {
        async_copy_b128(&As[bi][rS      ][kc], gA0 + k0);
        async_copy_b128(&As[bi][rS +  64][kc], gA1 + k0);
        async_copy_b128(&As[bi][rS + 128][kc], gA2 + k0);
        async_copy_b128(&As[bi][rS + 192][kc], gA3 + k0);
        async_copy_b128(&Bs[bi][rS      ][kc], gB0 + k0);
        async_copy_b128(&Bs[bi][rS +  64][kc], gB1 + k0);
    };

    v8f acc[4][4];
    for (int mt = 0; mt < 4; ++mt)
        for (int nt = 0; nt < 4; ++nt)
            acc[mt][nt] = (v8f){0.f,0.f,0.f,0.f,0.f,0.f,0.f,0.f};

    auto compute_tile = [&](int bi) {
        // ---- build WMMA fragments (CDNA5 wave32 layouts) ----
        BF16Frag afr[4];
        for (int mt = 0; mt < 4; ++mt) {
            const __bf16* ap = &As[bi][wm * 64 + mt * 16 + l15][hi * 8];
            afr[mt].h[0] = *reinterpret_cast<const v8bf*>(ap);        // K hi*8..+7
            afr[mt].h[1] = *reinterpret_cast<const v8bf*>(ap + 16);   // K 16+hi*8..
        }
        BF16Frag bfr[4];
        for (int nt = 0; nt < 4; ++nt) {
            const __bf16* bp = &Bs[bi][wn * 64 + nt * 16 + l15][hi * 8];
            bfr[nt].h[0] = *reinterpret_cast<const v8bf*>(bp);        // K hi*8..+7
            bfr[nt].h[1] = *reinterpret_cast<const v8bf*>(bp + 16);   // K 16+hi*8..
        }
        for (int mt = 0; mt < 4; ++mt)
            for (int nt = 0; nt < 4; ++nt)
                acc[mt][nt] = __builtin_amdgcn_wmma_f32_16x16x32_bf16(
                    false, afr[mt].f, false, bfr[nt].f,
                    (short)0, acc[mt][nt], false, false);
    };

    const int nk = K >> 5;               // number of 32-wide K tiles
    issue_tile(0, 0);

    int it = 0;
    for (; it < nk - 1; ++it) {
        issue_tile((it + 1) << 5, (it & 1) ^ 1);
        wait_asynccnt<6>();              // current tile's 6 copies done (in order)
        __syncthreads();
        compute_tile(it & 1);
        __syncthreads();                 // readers done before buffer refill
    }
    wait_asynccnt<0>();
    __syncthreads();
    compute_tile(it & 1);

    // ---- epilogue ----
    for (int mt = 0; mt < 4; ++mt) {
        for (int nt = 0; nt < 4; ++nt) {
            const int col = n0 + wn * 64 + nt * 16 + l15;
            float bv = 0.f;
            if (HAS_BIAS) bv = bias[col];
            for (int r = 0; r < 8; ++r) {
                const int row = m0 + wm * 64 + mt * 16 + hi * 8 + r;
                float v = acc[mt][nt][r] + bv;
                if (GELU_ACT) v = 0.5f * v * (1.f + erff(v * 0.70710678118654752f));
                if (HAS_RES)  v += res[(size_t)row * N + col];
                const size_t o = (size_t)row * N + col;
                if (OUT_BF16) outB[o] = (__bf16)v;
                else          outF[o] = v;
            }
        }
    }
}

// ---------------------------------------------------------------------------
// RMSNorm: one block per row of D=1024, writes bf16
// ---------------------------------------------------------------------------
__global__ __launch_bounds__(256) void k_rmsnorm_bf16(
    const float* __restrict__ x, const float* __restrict__ w,
    __bf16* __restrict__ out, int D)
{
    __shared__ float red[256];
    const size_t row = blockIdx.x;
    const float* xr = x + row * D;
    float s = 0.f;
    for (int j = threadIdx.x; j < D; j += 256) { float v = xr[j]; s += v * v; }
    red[threadIdx.x] = s; __syncthreads();
    for (int o = 128; o > 0; o >>= 1) {
        if (threadIdx.x < o) red[threadIdx.x] += red[threadIdx.x + o];
        __syncthreads();
    }
    const float rinv = rsqrtf(red[0] / (float)D + 1e-6f);
    __bf16* orow = out + row * D;
    for (int j = threadIdx.x; j < D; j += 256)
        orow[j] = (__bf16)(xr[j] * rinv * w[j]);
}

// ---------------------------------------------------------------------------
// Causal softmax over one row (scale folded in), fp32 scores -> bf16 probs
// ---------------------------------------------------------------------------
__global__ __launch_bounds__(256) void k_softmax_causal(
    const float* __restrict__ s, __bf16* __restrict__ p, int T, float scale)
{
    __shared__ float red[256];
    const int row = blockIdx.x;
    const float* sr = s + (size_t)row * T;
    __bf16* pr = p + (size_t)row * T;
    const int n = row + 1;

    float mx = -3.0e38f;
    for (int j = threadIdx.x; j < n; j += 256) mx = fmaxf(mx, sr[j] * scale);
    red[threadIdx.x] = mx; __syncthreads();
    for (int o = 128; o > 0; o >>= 1) {
        if (threadIdx.x < o) red[threadIdx.x] = fmaxf(red[threadIdx.x], red[threadIdx.x + o]);
        __syncthreads();
    }
    const float m = red[0]; __syncthreads();

    float sum = 0.f;
    for (int j = threadIdx.x; j < n; j += 256) sum += expf(sr[j] * scale - m);
    red[threadIdx.x] = sum; __syncthreads();
    for (int o = 128; o > 0; o >>= 1) {
        if (threadIdx.x < o) red[threadIdx.x] += red[threadIdx.x + o];
        __syncthreads();
    }
    const float inv = 1.f / red[0];

    for (int j = threadIdx.x; j < T; j += 256)
        pr[j] = (j < n) ? (__bf16)(expf(sr[j] * scale - m) * inv) : (__bf16)0.f;
}

// ---------------------------------------------------------------------------
// fp32 [K][N] -> bf16 [N][K] transpose-convert (weights)
// ---------------------------------------------------------------------------
__global__ __launch_bounds__(256) void k_transpose_f32_to_bf16(
    const float* __restrict__ in, __bf16* __restrict__ out, int K, int N)
{
    __shared__ float tile[32][33];
    const int k0 = blockIdx.y * 32, n0 = blockIdx.x * 32;
    const int tx = threadIdx.x, ty = threadIdx.y;   // 32 x 8
    for (int i = 0; i < 32; i += 8)
        tile[ty + i][tx] = in[(size_t)(k0 + ty + i) * N + n0 + tx];
    __syncthreads();
    for (int i = 0; i < 32; i += 8)
        out[(size_t)(n0 + ty + i) * K + k0 + tx] = (__bf16)tile[tx][ty + i];
}

// ---------------------------------------------------------------------------
// bf16 [R][C] -> bf16 [C][R] transpose, batched by blockIdx.z (for V^T)
// ---------------------------------------------------------------------------
__global__ __launch_bounds__(256) void k_transpose_bf16(
    const __bf16* __restrict__ in, __bf16* __restrict__ out, int R, int C)
{
    __shared__ __bf16 tile[32][33];
    const size_t boff = (size_t)blockIdx.z * R * C;
    const int r0 = blockIdx.y * 32, c0 = blockIdx.x * 32;
    const int tx = threadIdx.x, ty = threadIdx.y;   // 32 x 8
    for (int i = 0; i < 32; i += 8)
        tile[ty + i][tx] = in[boff + (size_t)(r0 + ty + i) * C + c0 + tx];
    __syncthreads();
    for (int i = 0; i < 32; i += 8)
        out[boff + (size_t)(c0 + ty + i) * R + r0 + tx] = tile[tx][ty + i];
}

// ---------------------------------------------------------------------------
// Orchestration
// ---------------------------------------------------------------------------
extern "C" void kernel_launch(void* const* d_in, const int* in_sizes, int n_in,
                              void* d_out, int out_size, void* d_ws, size_t ws_size,
                              hipStream_t stream)
{
    const int B = 8, T = 2048, D = 1024, H = 4096;
    const int M = B * T;                      // 16384 rows
    const float inv_sqrt_d = 0.03125f;        // 1/sqrt(1024)

    const float* x   = (const float*)d_in[0];
    const float* anw = (const float*)d_in[1];
    const float* mnw = (const float*)d_in[2];
    const float* wq  = (const float*)d_in[3];  const float* bq = (const float*)d_in[4];
    const float* wk  = (const float*)d_in[5];  const float* bk = (const float*)d_in[6];
    const float* wv  = (const float*)d_in[7];  const float* bv = (const float*)d_in[8];
    const float* wo  = (const float*)d_in[9];  const float* bo = (const float*)d_in[10];
    const float* w1  = (const float*)d_in[11]; const float* b1 = (const float*)d_in[12];
    const float* w2  = (const float*)d_in[13]; const float* b2 = (const float*)d_in[14];
    float* out = (float*)d_out;

    // workspace partitioning (~352 MB; scores/probs/g reused per batch)
    char* p = (char*)d_ws;
    auto alloc = [&](size_t bytes) { char* r = p; p += (bytes + 255) & ~size_t(255); return r; };
    __bf16* wqT  = (__bf16*)alloc((size_t)D * D * 2);
    __bf16* wkT  = (__bf16*)alloc((size_t)D * D * 2);
    __bf16* wvT  = (__bf16*)alloc((size_t)D * D * 2);
    __bf16* woT  = (__bf16*)alloc((size_t)D * D * 2);
    __bf16* w1T  = (__bf16*)alloc((size_t)D * H * 2);
    __bf16* w2T  = (__bf16*)alloc((size_t)H * D * 2);
    __bf16* xn   = (__bf16*)alloc((size_t)M * D * 2);
    __bf16* qb   = (__bf16*)alloc((size_t)M * D * 2);
    __bf16* kb   = (__bf16*)alloc((size_t)M * D * 2);
    __bf16* vb   = (__bf16*)alloc((size_t)M * D * 2);
    __bf16* vT   = (__bf16*)alloc((size_t)M * D * 2);
    __bf16* attn = (__bf16*)alloc((size_t)M * D * 2);
    float*  x1   = (float*) alloc((size_t)M * D * 4);
    __bf16* hb   = (__bf16*)alloc((size_t)M * D * 2);
    float*  sc   = (float*) alloc((size_t)T * T * 4);   // per-batch scores
    __bf16* pb   = (__bf16*)alloc((size_t)T * T * 2);   // per-batch probs
    __bf16* gb   = (__bf16*)alloc((size_t)T * H * 2);   // per-batch gelu out

    const dim3 tb(32, 8);

    // 1) weight transpose+convert to bf16 [N][K]
    k_transpose_f32_to_bf16<<<dim3(D/32, D/32), tb, 0, stream>>>(wq, wqT, D, D);
    k_transpose_f32_to_bf16<<<dim3(D/32, D/32), tb, 0, stream>>>(wk, wkT, D, D);
    k_transpose_f32_to_bf16<<<dim3(D/32, D/32), tb, 0, stream>>>(wv, wvT, D, D);
    k_transpose_f32_to_bf16<<<dim3(D/32, D/32), tb, 0, stream>>>(wo, woT, D, D);
    k_transpose_f32_to_bf16<<<dim3(H/32, D/32), tb, 0, stream>>>(w1, w1T, D, H);
    k_transpose_f32_to_bf16<<<dim3(D/32, H/32), tb, 0, stream>>>(w2, w2T, H, D);

    // 2) rmsnorm(x) -> xn (bf16)
    k_rmsnorm_bf16<<<dim3(M), 256, 0, stream>>>(x, anw, xn, D);

    // 3) Q/K/V projections (bf16 out)
    k_gemm_bf16<true,false,false,true><<<dim3(D/128, M/256), 256, 0, stream>>>(
        xn, wqT, bq, nullptr, nullptr, qb, M, D, D);
    k_gemm_bf16<true,false,false,true><<<dim3(D/128, M/256), 256, 0, stream>>>(
        xn, wkT, bk, nullptr, nullptr, kb, M, D, D);
    k_gemm_bf16<true,false,false,true><<<dim3(D/128, M/256), 256, 0, stream>>>(
        xn, wvT, bv, nullptr, nullptr, vb, M, D, D);

    // 4) V^T per batch: [T][D] -> [D][T]
    k_transpose_bf16<<<dim3(D/32, T/32, B), tb, 0, stream>>>(vb, vT, T, D);

    // 5) attention per batch (scores buffer reused)
    for (int b = 0; b < B; ++b) {
        const __bf16* qB = qb + (size_t)b * T * D;
        const __bf16* kB = kb + (size_t)b * T * D;   // [T][D] == Bt[N=T][K=D]
        const __bf16* vTB = vT + (size_t)b * T * D;  // [D][T] == Bt[N=D][K=T]
        __bf16* aB = attn + (size_t)b * T * D;
        // scores = q @ k^T  (fp32)
        k_gemm_bf16<false,false,false,false><<<dim3(T/128, T/256), 256, 0, stream>>>(
            qB, kB, nullptr, nullptr, sc, nullptr, T, T, D);
        // causal softmax with 1/sqrt(d) scale -> bf16 probs
        k_softmax_causal<<<dim3(T), 256, 0, stream>>>(sc, pb, T, inv_sqrt_d);
        // attn = probs @ v  (bf16)
        k_gemm_bf16<false,false,false,true><<<dim3(D/128, T/256), 256, 0, stream>>>(
            pb, vTB, nullptr, nullptr, nullptr, aB, T, D, T);
    }

    // 6) x1 = x + attn @ wo + bo  (fp32)
    k_gemm_bf16<true,true,false,false><<<dim3(D/128, M/256), 256, 0, stream>>>(
        attn, woT, bo, x, x1, nullptr, M, D, D);

    // 7) h = rmsnorm(x1) (bf16)
    k_rmsnorm_bf16<<<dim3(M), 256, 0, stream>>>(x1, mnw, hb, D);

    // 8) MLP per batch (g buffer reused)
    for (int b = 0; b < B; ++b) {
        const __bf16* hB = hb + (size_t)b * T * D;
        const float*  rB = x1 + (size_t)b * T * D;
        float*        oB = out + (size_t)b * T * D;
        // g = gelu(h @ w1 + b1)  (bf16)
        k_gemm_bf16<true,false,true,true><<<dim3(H/128, T/256), 256, 0, stream>>>(
            hB, w1T, b1, nullptr, nullptr, gb, T, H, D);
        // out = x1 + g @ w2 + b2  (fp32)
        k_gemm_bf16<true,true,false,false><<<dim3(D/128, T/256), 256, 0, stream>>>(
            gb, w2T, b2, rB, oB, nullptr, T, D, H);
    }
    (void)in_sizes; (void)n_in; (void)out_size; (void)ws_size;
}
